// SeedSelection_29446295781588
// MI455X (gfx1250) — matrix operand order repaired and checked by reading, not compiled, not run on hardware
//
#include <hip/hip_runtime.h>
#include <hip/hip_bf16.h>
#include <stdint.h>

typedef __attribute__((ext_vector_type(16))) _Float16 v16h;
typedef __attribute__((ext_vector_type(8)))  _Float16 v8h;
typedef __attribute__((ext_vector_type(8)))  float    v8f;

#define Lc   6
#define Bc   4
#define Qc   900
#define Cc   256
#define Nn   5400          // L*Q valid nodes per image
#define NPc  5408          // padded to multiple of 32
#define Wc   169           // NPc/32 adjacency words per row
#define TT2c 169           // NPc/32 32x32 wave-tiles per dim
#define SEED_THR 0.5f
#define DUP_THR  0.8f

// ---------------- init: labels = i, compBest/argIdx = INT_MAX ----------------
__global__ void k_init(int* labels, int* compBest, int* argIdx) {
    int idx = blockIdx.x * blockDim.x + threadIdx.x;
    int total = Bc * NPc;
    if (idx < total) {
        labels[idx]   = idx % NPc;
        compBest[idx] = 0x7fffffff;
    }
    if (idx < Bc) argIdx[idx] = 0x7fffffff;
}

// ---------------- scores = sigmoid(logits); raw selection; per-image stats ----
__global__ void k_score(const float* __restrict__ logits, float* __restrict__ scores,
                        int* __restrict__ selW, int* __restrict__ cnt,
                        unsigned* __restrict__ maxBits) {
    int idx = blockIdx.x * blockDim.x + threadIdx.x;
    if (idx >= Bc * NPc) return;
    int b = idx / NPc, n = idx % NPc;
    if (n >= Nn) { selW[idx] = 0; return; }      // scores already 0 from memset
    int l = n / Qc, q = n % Qc;
    float lg = logits[((size_t)l * Bc + b) * Qc + q];
    float s  = 1.0f / (1.0f + __expf(-lg));
    scores[idx] = s;
    int sel = (s >= SEED_THR) ? 1 : 0;
    selW[idx] = sel;
    if (sel) atomicAdd(&cnt[b], 1);
    atomicMax(&maxBits[b], __float_as_uint(s));  // scores > 0 -> int-ordered
}

__global__ void k_argmin(const float* __restrict__ scores, const unsigned* __restrict__ maxBits,
                         int* __restrict__ argIdx) {
    int idx = blockIdx.x * blockDim.x + threadIdx.x;
    if (idx >= Bc * NPc) return;
    int b = idx / NPc, n = idx % NPc;
    if (n >= Nn) return;
    if (__float_as_uint(scores[idx]) == maxBits[b]) atomicMin(&argIdx[b], n);
}

__global__ void k_selfix(int* __restrict__ selW, const int* __restrict__ cnt,
                         const int* __restrict__ argIdx) {
    int idx = blockIdx.x * blockDim.x + threadIdx.x;
    if (idx >= Bc * NPc) return;
    int b = idx / NPc, n = idx % NPc;
    if (cnt[b] == 0) selW[idx] = (n == argIdx[b]) ? 1 : 0;  // argmax fallback
}

// ---------------- L2-normalize signatures -> f16, [B, NP, C] ----------------
__global__ void k_norm(const float* __restrict__ sig, _Float16* __restrict__ sn) {
    int wave = (blockIdx.x * blockDim.x + threadIdx.x) >> 5;
    int lane = threadIdx.x & 31;
    if (wave >= Bc * NPc) return;
    int b = wave / NPc, n = wave % NPc;
    _Float16* out = sn + ((size_t)b * NPc + n) * Cc;
    if (n >= Nn) {
        for (int c = lane; c < Cc; c += 32) out[c] = (_Float16)0.0f;
        return;
    }
    int l = n / Qc, q = n % Qc;
    const float* row = sig + (((size_t)l * Bc + b) * Qc + q) * (size_t)Cc;
    float v[8]; float ss = 0.0f;
    #pragma unroll
    for (int t = 0; t < 8; ++t) { v[t] = row[lane + t * 32]; ss += v[t] * v[t]; }
    #pragma unroll
    for (int off = 16; off; off >>= 1) ss += __shfl_xor(ss, off, 32);
    float r = rsqrtf(ss + 1e-12f);
    #pragma unroll
    for (int t = 0; t < 8; ++t) out[lane + t * 32] = (_Float16)(v[t] * r);
}

// ---------------- WMMA fragment loaders (per ISA VGPR layouts) ---------------
// A 16x32 f16: lanes 0-15 hold K={0..7,16..23}, lanes 16-31 hold K={8..15,24..31}
__device__ __forceinline__ v16h load_afrag(const _Float16* p, int hi) {
    v8h lo = *(const v8h*)(p + (hi ? 8 : 0));
    v8h hh = *(const v8h*)(p + (hi ? 8 : 0) + 16);
    v16h a;
    #pragma unroll
    for (int t = 0; t < 8; ++t) { a[t] = lo[t]; a[t + 8] = hh[t]; }
    return a;
}
// B 32x16 f16: lanes 0-15 hold K={0..7,16..23}, lanes 16-31 hold K={8..15,24..31}
__device__ __forceinline__ v16h load_bfrag(const _Float16* p, int hi) {
    v8h lo = *(const v8h*)(p + hi * 8);
    v8h hh = *(const v8h*)(p + hi * 8 + 16);
    v16h bm;
    #pragma unroll
    for (int t = 0; t < 8; ++t) { bm[t] = lo[t]; bm[t + 8] = hh[t]; }
    return bm;
}

// ---------------- WMMA pairwise cosine similarity -> adjacency bitset --------
// One wave per 32x32 output block (2x2 grid of 16x16 WMMA tiles).
// A/B fragment reuse doubles WMMA:VMEM ratio; each wave owns whole 32-bit
// adjacency words -> plain stores, no atomics.
__global__ void k_gemm(const _Float16* __restrict__ sn, const int* __restrict__ selW,
                       unsigned* __restrict__ adj) {
    int wave = (blockIdx.x * blockDim.x + threadIdx.x) >> 5;   // uniform per wave
    int lane = threadIdx.x & 31;
    if (wave >= Bc * TT2c * TT2c) return;                      // wave-uniform exit
    int b   = wave / (TT2c * TT2c);
    int rem = wave % (TT2c * TT2c);
    int tm  = (rem / TT2c) * 32;
    int tn  = (rem % TT2c) * 32;
    const _Float16* base = sn + (size_t)b * NPc * Cc;
    const int* selB = selW + (size_t)b * NPc;
    int hi = lane >> 4;            // lane half selects K sub-block
    int ls = lane & 15;
    const _Float16* aRow0 = base + (size_t)(tm + ls)      * Cc;
    const _Float16* aRow1 = base + (size_t)(tm + 16 + ls) * Cc;
    const _Float16* bRow0 = base + (size_t)(tn + ls)      * Cc;
    const _Float16* bRow1 = base + (size_t)(tn + 16 + ls) * Cc;
    v8f acc00 = {}, acc01 = {}, acc10 = {}, acc11 = {};
    #pragma unroll
    for (int k0 = 0; k0 < Cc; k0 += 32) {
        v16h a0 = load_afrag(aRow0 + k0, hi);
        v16h a1 = load_afrag(aRow1 + k0, hi);
        v16h b0 = load_bfrag(bRow0 + k0, hi);
        v16h b1 = load_bfrag(bRow1 + k0, hi);
        acc00 = __builtin_amdgcn_wmma_f32_16x16x32_f16(false, a0, false, b0, (short)0, acc00, false, false);
        acc01 = __builtin_amdgcn_wmma_f32_16x16x32_f16(false, a0, false, b1, (short)0, acc01, false, false);
        acc10 = __builtin_amdgcn_wmma_f32_16x16x32_f16(false, a1, false, b0, (short)0, acc10, false, false);
        acc11 = __builtin_amdgcn_wmma_f32_16x16x32_f16(false, a1, false, b1, (short)0, acc11, false, false);
    }
    // selection masks
    unsigned maskM0 = (unsigned)__ballot(selB[tm + ls]      != 0) & 0xFFFFu;
    unsigned maskM1 = (unsigned)__ballot(selB[tm + 16 + ls] != 0) & 0xFFFFu;
    int selN0 = selB[tn + ls];
    int selN1 = selB[tn + 16 + ls];
    int wordIdx = tn >> 5;                 // exactly one word per row for this wave
    size_t rowBase = (size_t)b * NPc;
    #pragma unroll
    for (int r = 0; r < 8; ++r) {
        // D layout: VGPR r -> (m=+r) lanes 0-15 ; (m=+r+8) lanes 16-31
        int sM0 = (maskM0 >> (r + hi * 8)) & 1;
        int sM1 = (maskM1 >> (r + hi * 8)) & 1;
        unsigned b00 = (unsigned)__ballot(sM0 && selN0 && (acc00[r] >= DUP_THR));
        unsigned b01 = (unsigned)__ballot(sM0 && selN1 && (acc01[r] >= DUP_THR));
        unsigned b10 = (unsigned)__ballot(sM1 && selN0 && (acc10[r] >= DUP_THR));
        unsigned b11 = (unsigned)__ballot(sM1 && selN1 && (acc11[r] >= DUP_THR));
        if (lane == 0) {
            // merge sub-tile ballots into full 32-column row words
            unsigned w0lo = (b00 & 0xFFFFu) | ((b01 & 0xFFFFu) << 16);  // row tm+r
            unsigned w0hi = (b00 >> 16)     | (b01 & 0xFFFF0000u);      // row tm+r+8
            unsigned w1lo = (b10 & 0xFFFFu) | ((b11 & 0xFFFFu) << 16);  // row tm+r+16
            unsigned w1hi = (b10 >> 16)     | (b11 & 0xFFFF0000u);      // row tm+r+24
            adj[(rowBase + tm + r)      * Wc + wordIdx] = w0lo;
            adj[(rowBase + tm + r + 8)  * Wc + wordIdx] = w0hi;
            adj[(rowBase + tm + r + 16) * Wc + wordIdx] = w1lo;
            adj[(rowBase + tm + r + 24) * Wc + wordIdx] = w1hi;
        }
    }
}

// ---------------- min-label propagation over bitset (one wave per node) ------
__global__ void k_prop(const unsigned* __restrict__ adj, const int* __restrict__ labels,
                       int* __restrict__ labelsOut) {
    int wave = (blockIdx.x * blockDim.x + threadIdx.x) >> 5;
    int lane = threadIdx.x & 31;
    if (wave >= Bc * NPc) return;
    int b = wave / NPc, i = wave % NPc;
    const int* lab = labels + (size_t)b * NPc;
    int l = lab[i];
    const unsigned* row = adj + ((size_t)b * NPc + i) * Wc;
    for (int w = lane; w < Wc; w += 32) {
        unsigned bits = row[w];
        while (bits) {
            int j = w * 32 + __builtin_ctz(bits);
            bits &= bits - 1;
            int lj = lab[j];
            l = (lj < l) ? lj : l;
        }
    }
    #pragma unroll
    for (int off = 16; off; off >>= 1) { int o = __shfl_xor(l, off, 32); l = (o < l) ? o : l; }
    if (lane == 0) labelsOut[(size_t)b * NPc + i] = l;
}

__global__ void k_jump(const int* __restrict__ labelsIn, int* __restrict__ labels) {
    int idx = blockIdx.x * blockDim.x + threadIdx.x;
    if (idx >= Bc * NPc) return;
    int b = idx / NPc;
    int l = labelsIn[idx];
    int l2 = labelsIn[(size_t)b * NPc + l];                   // pointer jumping
    labels[idx] = (l2 < l) ? l2 : l;
}

// ---------------- per-component max score / best index / seed flags ----------
__global__ void k_compmax(const int* __restrict__ selW, const float* __restrict__ scores,
                          const int* __restrict__ labels, unsigned* __restrict__ compMax) {
    int idx = blockIdx.x * blockDim.x + threadIdx.x;
    if (idx >= Bc * NPc) return;
    if (!selW[idx]) return;
    int b = idx / NPc;
    atomicMax(&compMax[(size_t)b * NPc + labels[idx]], __float_as_uint(scores[idx]));
}

__global__ void k_best(const int* __restrict__ selW, const float* __restrict__ scores,
                       const int* __restrict__ labels, const unsigned* __restrict__ compMax,
                       int* __restrict__ compBest) {
    int idx = blockIdx.x * blockDim.x + threadIdx.x;
    if (idx >= Bc * NPc) return;
    if (!selW[idx]) return;
    int b = idx / NPc, n = idx % NPc;
    size_t root = (size_t)b * NPc + labels[idx];
    if (scores[idx] >= __uint_as_float(compMax[root])) atomicMin(&compBest[root], n);
}

__global__ void k_seed(const int* __restrict__ selW, const int* __restrict__ labels,
                       const int* __restrict__ compBest, int* __restrict__ isSeed,
                       int* __restrict__ flag) {
    int idx = blockIdx.x * blockDim.x + threadIdx.x;
    if (idx >= Bc * NPc) return;
    int b = idx / NPc, n = idx % NPc;
    size_t root = (size_t)b * NPc + labels[idx];
    int s = selW[idx] && (n == compBest[root]);
    isSeed[idx] = s;
    if (s) flag[root] = 1;                                    // unique writer per root
}

// ---------------- exclusive prefix over seed-root flags (rank by root) -------
__global__ void k_scan(const int* __restrict__ flag, int* __restrict__ prefix) {
    __shared__ int s[NPc];
    int b = blockIdx.x;
    for (int i = threadIdx.x; i < NPc; i += blockDim.x) s[i] = flag[(size_t)b * NPc + i];
    __syncthreads();
    if (threadIdx.x == 0) {
        int acc = 0;
        for (int i = 0; i < NPc; ++i) { int f = s[i]; s[i] = acc; acc += f; }
    }
    __syncthreads();
    for (int i = threadIdx.x; i < NPc; i += blockDim.x) prefix[(size_t)b * NPc + i] = s[i];
}

// ---------------- scatter seeds to compacted output positions ----------------
__global__ void k_scatter(const float* __restrict__ sig, const float* __restrict__ scores,
                          const int* __restrict__ labels, const int* __restrict__ isSeed,
                          const int* __restrict__ prefix,
                          float* __restrict__ outSig, float* __restrict__ outMask,
                          float* __restrict__ outSc) {
    int node = blockIdx.x;                 // B*NP blocks of 64 threads
    int b = node / NPc, n = node % NPc;
    if (!isSeed[node]) return;
    int pos = prefix[(size_t)b * NPc + labels[node]];
    if (pos >= Nn) return;
    if (threadIdx.x == 0) {
        outMask[(size_t)b * Nn + pos] = 1.0f;
        outSc[(size_t)b * Nn + pos]   = scores[node];
    }
    int l = n / Qc, q = n % Qc;
    const float* src = sig + (((size_t)l * Bc + b) * Qc + q) * (size_t)Cc;
    float* dst = outSig + ((size_t)b * Nn + pos) * Cc;
    for (int c = threadIdx.x; c < Cc; c += 64) dst[c] = src[c];
}

// ============================ launcher =======================================
extern "C" void kernel_launch(void* const* d_in, const int* in_sizes, int n_in,
                              void* d_out, int out_size, void* d_ws, size_t ws_size,
                              hipStream_t stream) {
    const float* sig    = (const float*)d_in[0];   // [L,B,Q,C] f32
    const float* logits = (const float*)d_in[1];   // [L,B,Q]   f32

    // ---- carve workspace (256B aligned) ----
    uint8_t* ws = (uint8_t*)d_ws;
    size_t off = 0;
    auto carve = [&](size_t bytes) -> uint8_t* {
        uint8_t* p = ws + off;
        off = (off + bytes + 255) & ~(size_t)255;
        return p;
    };
    const size_t nodes = (size_t)Bc * NPc;
    _Float16* sn       = (_Float16*)carve(nodes * Cc * sizeof(_Float16));
    float*    scores   = (float*)   carve(nodes * sizeof(float));
    int*      selW     = (int*)     carve(nodes * sizeof(int));
    int*      labels   = (int*)     carve(nodes * sizeof(int));
    int*      labelsB  = (int*)     carve(nodes * sizeof(int));
    unsigned* adj      = (unsigned*)carve(nodes * Wc * sizeof(unsigned));
    unsigned* compMax  = (unsigned*)carve(nodes * sizeof(unsigned));
    int*      compBest = (int*)     carve(nodes * sizeof(int));
    int*      isSeed   = (int*)     carve(nodes * sizeof(int));
    int*      flag     = (int*)     carve(nodes * sizeof(int));
    int*      prefix   = (int*)     carve(nodes * sizeof(int));
    int*      cnt      = (int*)     carve(Bc * sizeof(int));
    unsigned* maxBits  = (unsigned*)carve(Bc * sizeof(unsigned));
    int*      argIdx   = (int*)     carve(Bc * sizeof(int));
    size_t used = off;
    (void)ws_size; (void)n_in; (void)in_sizes;

    // fresh state every call (deterministic)
    hipMemsetAsync(d_ws, 0, used, stream);
    hipMemsetAsync(d_out, 0, (size_t)out_size * sizeof(float), stream);

    const int thr256 = 256;
    const int blkNodes = (int)((nodes + thr256 - 1) / thr256);   // thread-per-node
    const int blkWaves = (int)(nodes / 8);                       // wave-per-node, 8 waves/blk

    k_init  <<<blkNodes, thr256, 0, stream>>>(labels, compBest, argIdx);
    k_score <<<blkNodes, thr256, 0, stream>>>(logits, scores, selW, cnt, maxBits);
    k_argmin<<<blkNodes, thr256, 0, stream>>>(scores, maxBits, argIdx);
    k_selfix<<<blkNodes, thr256, 0, stream>>>(selW, cnt, argIdx);
    k_norm  <<<blkWaves, thr256, 0, stream>>>(sig, sn);

    const int totalWaveTiles = Bc * TT2c * TT2c;                 // 114,244
    const int gemmBlocks = (totalWaveTiles + 7) / 8;             // 8 waves / block
    k_gemm  <<<gemmBlocks, thr256, 0, stream>>>(sn, selW, adj);

    for (int it = 0; it < 16; ++it) {
        k_prop<<<blkWaves, thr256, 0, stream>>>(adj, labels, labelsB);
        k_jump<<<blkNodes, thr256, 0, stream>>>(labelsB, labels);
    }

    k_compmax<<<blkNodes, thr256, 0, stream>>>(selW, scores, labels, compMax);
    k_best   <<<blkNodes, thr256, 0, stream>>>(selW, scores, labels, compMax, compBest);
    k_seed   <<<blkNodes, thr256, 0, stream>>>(selW, labels, compBest, isSeed, flag);
    k_scan   <<<Bc, 1024, 0, stream>>>(flag, prefix);

    float* outSig  = (float*)d_out;
    float* outMask = outSig + (size_t)Bc * Nn * Cc;
    float* outSc   = outMask + (size_t)Bc * Nn;
    k_scatter<<<(int)nodes, 64, 0, stream>>>(sig, scores, labels, isSeed, prefix,
                                             outSig, outMask, outSc);
}
